// CrossAttention_16758962389441
// MI455X (gfx1250) — compile-verified, matrix operand order checked by reading, not run on hardware
//
#include <hip/hip_runtime.h>

// ---------------------------------------------------------------------------
// CrossAttention for MI455X (gfx1250, wave32, WMMA).
//
// Roofline: 40.5 GFLOP total (two 32768x512x512 GEMMs dominate), ~140 MB
// mandatory HBM traffic (~6 us @ 23.3 TB/s). Compute-leaning => route all
// matmuls through v_wmma_f32_16x16x32_bf16 (bf16 inputs, f32 accumulate),
// software-pipeline fragment loads against WMMAs, and keep fp32->bf16
// conversion out of the hot GEMM loops (dedicated memory-bound pass).
// ---------------------------------------------------------------------------

#define DEVINL __device__ __forceinline__

typedef __bf16 bf16_t;
typedef bf16_t v16bf __attribute__((ext_vector_type(16)));
typedef bf16_t v8bf  __attribute__((ext_vector_type(8)));
typedef float  v8f   __attribute__((ext_vector_type(8)));
typedef float  v4f   __attribute__((ext_vector_type(4)));
typedef unsigned int v4u __attribute__((ext_vector_type(4)));

DEVINL unsigned short f32_to_bf16_bits(float f) {
  unsigned int u = __float_as_uint(f);
  u += 0x7FFFu + ((u >> 16) & 1u);   // round-to-nearest-even
  return (unsigned short)(u >> 16);
}

// Pack two fp32 into two bf16 (one dword). Prefer the HW packed convert.
#if __has_builtin(__builtin_amdgcn_cvt_pk_bf16_f32)
DEVINL unsigned int pack2_bf16(float lo, float hi) {
  auto v = __builtin_amdgcn_cvt_pk_bf16_f32(lo, hi);
  static_assert(sizeof(v) == 4, "unexpected cvt_pk_bf16_f32 result size");
  return __builtin_bit_cast(unsigned int, v);
}
#else
DEVINL unsigned int pack2_bf16(float lo, float hi) {
  return (unsigned int)f32_to_bf16_bits(lo) |
         ((unsigned int)f32_to_bf16_bits(hi) << 16);
}
#endif

// 16x32 bf16 A-operand fragment from row-major memory (also used for the B
// operand by pointing at B^T, which has the identical lane layout).
// ISA layout: lane<16 -> row=lane, K in {0..7} (VGPR0-3) and {16..23} (VGPR4-7)
//             lane>=16 -> row=lane-16, K in {8..15} and {24..31}
DEVINL v16bf load_frag_bf16(const unsigned short* base, int ldm, int lane) {
  const int half = lane >> 4, r = lane & 15;
  const unsigned short* p = base + (size_t)r * ldm + half * 8;
  v8bf lo = *reinterpret_cast<const v8bf*>(p);       // K = half*8 + 0..7
  v8bf hi = *reinterpret_cast<const v8bf*>(p + 16);  // K = 16 + half*8 + 0..7
  v16bf out;
#pragma unroll
  for (int i = 0; i < 8; ++i) { out[i] = lo[i]; out[i + 8] = hi[i]; }
  return out;
}

// A fragment from row-major fp32 with on-the-fly bf16 convert and a
// BRANCHLESS row guard: rows >= rowsValid read a safe in-bounds row and are
// masked to zero with an AND (no exec divergence in the hot loop).
DEVINL v16bf load_fragA_f32cvt_mask(const float* base, int ldm, int lane,
                                    int rowsValid) {
  const int half = lane >> 4, rr = lane & 15;
  const int rSafe = (rr < rowsValid) ? rr : 0;
  const unsigned int msk = (rr < rowsValid) ? 0xFFFFFFFFu : 0u;
  const float* p = base + (size_t)rSafe * ldm + half * 8;
  v4f a0 = *reinterpret_cast<const v4f*>(p);
  v4f a1 = *reinterpret_cast<const v4f*>(p + 4);
  v4f b0 = *reinterpret_cast<const v4f*>(p + 16);
  v4f b1 = *reinterpret_cast<const v4f*>(p + 20);
  union { v16bf v; unsigned int u[8]; } r;
  r.u[0] = pack2_bf16(a0[0], a0[1]) & msk; r.u[1] = pack2_bf16(a0[2], a0[3]) & msk;
  r.u[2] = pack2_bf16(a1[0], a1[1]) & msk; r.u[3] = pack2_bf16(a1[2], a1[3]) & msk;
  r.u[4] = pack2_bf16(b0[0], b0[1]) & msk; r.u[5] = pack2_bf16(b0[2], b0[3]) & msk;
  r.u[6] = pack2_bf16(b1[0], b1[1]) & msk; r.u[7] = pack2_bf16(b1[2], b1[3]) & msk;
  return r.v;
}

DEVINL v8f wmma_bf16(v16bf a, v16bf b, v8f c) {
  // (neg_a, A, neg_b, B, c_mod, C, reuse_a, reuse_b)
  return __builtin_amdgcn_wmma_f32_16x16x32_bf16(false, a, false, b, (short)0, c,
                                                 false, false);
}

// ---------------------------------------------------------------------------
// Weight convert + transpose: in [K][N] fp32 -> out [N][K] bf16.
// ---------------------------------------------------------------------------
__global__ __launch_bounds__(256)
void convert_transpose_kernel(const float* __restrict__ in,
                              unsigned short* __restrict__ out, int K, int N) {
  int idx = blockIdx.x * 256 + threadIdx.x;
  if (idx >= K * N) return;
  int k = idx / N, n = idx - k * N;
  out[(size_t)n * K + k] = f32_to_bf16_bits(in[idx]);
}

// ---------------------------------------------------------------------------
// Bulk fp32 -> bf16 convert (memory-bound): 8 floats per thread, 16B store.
// n8 = element count / 8 (divisible in our use).
// ---------------------------------------------------------------------------
__global__ __launch_bounds__(256)
void f32_to_bf16_bulk_kernel(const float* __restrict__ in,
                             unsigned int* __restrict__ out, int n8) {
  int idx = blockIdx.x * 256 + threadIdx.x;
  if (idx >= n8) return;
  const float* p = in + (size_t)idx * 8;
  v4f a = *reinterpret_cast<const v4f*>(p);
  v4f b = *reinterpret_cast<const v4f*>(p + 4);
  v4u r;
  r[0] = pack2_bf16(a[0], a[1]);
  r[1] = pack2_bf16(a[2], a[3]);
  r[2] = pack2_bf16(b[0], b[1]);
  r[3] = pack2_bf16(b[2], b[3]);
  *reinterpret_cast<v4u*>(out + (size_t)idx * 4) = r;
}

// ---------------------------------------------------------------------------
// GEMM: C[rows][512] = A[rows][512](bf16) @ W, W given as WT[512][512] bf16.
// Wave = 32x64 output tile (8 accumulators; B fragments reused by both row
// halves -> 1.5 b128 loads per WMMA). Block = 8 waves = 256 rows.
// Software-pipelined: step k+1 fragments are in flight during step k WMMAs.
// grid = (rows/256, 8).
// ---------------------------------------------------------------------------
template <bool ADD_BIAS, bool OUT_BF16>
__global__ __launch_bounds__(256)
void gemm_bf16_wt(const unsigned short* __restrict__ A,
                  const unsigned short* __restrict__ WT,
                  const float* __restrict__ bias, void* __restrict__ Cvoid) {
  constexpr int K = 512;
  constexpr int KSTEPS = K / 32;
  const int lane = threadIdx.x & 31;
  const int wave = threadIdx.x >> 5;
  const int row0 = (blockIdx.x * 8 + wave) * 32;
  const int n0 = blockIdx.y * 64;

  const unsigned short* A0 = A + (size_t)row0 * K;
  const unsigned short* A1 = A0 + (size_t)16 * K;
  const unsigned short* B0 = WT + (size_t)n0 * K;

  v16bf a_cur[2], b_cur[4];
  a_cur[0] = load_frag_bf16(A0, K, lane);
  a_cur[1] = load_frag_bf16(A1, K, lane);
#pragma unroll
  for (int t = 0; t < 4; ++t)
    b_cur[t] = load_frag_bf16(B0 + (size_t)t * 16 * K, K, lane);

  v8f acc[8] = {};
#pragma unroll 2
  for (int kk = 0; kk < KSTEPS - 1; ++kk) {
    const int ko = (kk + 1) * 32;
    if (kk + 2 < KSTEPS) {  // keep the A stream ahead in L2/WGP$
      __builtin_prefetch(A0 + (kk + 2) * 32 + (size_t)(lane & 15) * K, 0, 3);
      __builtin_prefetch(A1 + (kk + 2) * 32 + (size_t)(lane & 15) * K, 0, 3);
    }
    v16bf a_nxt[2], b_nxt[4];
    a_nxt[0] = load_frag_bf16(A0 + ko, K, lane);
    a_nxt[1] = load_frag_bf16(A1 + ko, K, lane);
#pragma unroll
    for (int t = 0; t < 4; ++t)
      b_nxt[t] = load_frag_bf16(B0 + (size_t)t * 16 * K + ko, K, lane);
#pragma unroll
    for (int t = 0; t < 4; ++t) {
      acc[t]     = wmma_bf16(a_cur[0], b_cur[t], acc[t]);
      acc[t + 4] = wmma_bf16(a_cur[1], b_cur[t], acc[t + 4]);
    }
    a_cur[0] = a_nxt[0];
    a_cur[1] = a_nxt[1];
#pragma unroll
    for (int t = 0; t < 4; ++t) b_cur[t] = b_nxt[t];
  }
#pragma unroll
  for (int t = 0; t < 4; ++t) {
    acc[t]     = wmma_bf16(a_cur[0], b_cur[t], acc[t]);
    acc[t + 4] = wmma_bf16(a_cur[1], b_cur[t], acc[t + 4]);
  }

  // D layout: VGPR i -> row i (lanes 0-15) / row i+8 (lanes 16-31); lane = col.
  const int col = lane & 15;
  const int rsel = (lane >> 4) * 8;
#pragma unroll
  for (int half = 0; half < 2; ++half) {
#pragma unroll
    for (int t = 0; t < 4; ++t) {
      float bv = 0.0f;
      if constexpr (ADD_BIAS) bv = bias[n0 + t * 16 + col];
      const v8f av = acc[half * 4 + t];
#pragma unroll
      for (int i = 0; i < 8; ++i) {
        size_t row = (size_t)row0 + half * 16 + rsel + i;
        int c = n0 + t * 16 + col;
        if constexpr (OUT_BF16)
          ((unsigned short*)Cvoid)[row * 512 + c] = f32_to_bf16_bits(av[i]);
        else
          ((float*)Cvoid)[row * 512 + c] = av[i] + bv;
      }
    }
  }
}

// ---------------------------------------------------------------------------
// K/V projection. context [B][77][768] fp32 -> per (b,h):
//   Kbuf [96][64] bf16 (row-major over m), VTbuf [64][96] bf16 (transposed).
// Rows m>=77 zero-filled (softmax masks them anyway; zeros avoid NaN).
// Only 768 waves run (low occupancy), so this kernel is latency-bound:
// the k-loop is software-pipelined like the big GEMM and the row guard is
// branchless (clamped address + AND mask) to keep the loop stall-free.
// Block = 8 waves (one per head); grid = (B*6 m-tiles, 2 for K|V).
// ---------------------------------------------------------------------------
__global__ __launch_bounds__(256)
void kv_proj_kernel(const float* __restrict__ context,
                    const unsigned short* __restrict__ WkT,
                    const unsigned short* __restrict__ WvT,
                    unsigned short* __restrict__ Kbuf,
                    unsigned short* __restrict__ VTbuf) {
  constexpr int KSTEPS = 768 / 32;
  const int lane = threadIdx.x & 31;
  const int h = threadIdx.x >> 5;
  const int bm = blockIdx.x;
  const int b = bm / 6, mt = bm - b * 6;
  const bool isV = blockIdx.y != 0;
  const int m0 = mt * 16;

  const unsigned short* WT = isV ? WvT : WkT;
  v8f acc[4] = {};
  const int rowsValid = (m0 < 77) ? ((77 - m0 < 16) ? 77 - m0 : 16) : 0;
  if (rowsValid > 0) {  // wave-uniform branch (tile 5 is pure zero-fill)
    const float* A0 = context + ((size_t)b * 77 + m0) * 768;
    const unsigned short* W0 = WT + (size_t)(h * 64) * 768;

    v16bf a_cur = load_fragA_f32cvt_mask(A0, 768, lane, rowsValid);
    v16bf w_cur[4];
#pragma unroll
    for (int t = 0; t < 4; ++t)
      w_cur[t] = load_frag_bf16(W0 + (size_t)t * 16 * 768, 768, lane);

#pragma unroll 2
    for (int kk = 0; kk < KSTEPS - 1; ++kk) {
      const int ko = (kk + 1) * 32;
      v16bf a_nxt = load_fragA_f32cvt_mask(A0 + ko, 768, lane, rowsValid);
      v16bf w_nxt[4];
#pragma unroll
      for (int t = 0; t < 4; ++t)
        w_nxt[t] = load_frag_bf16(W0 + (size_t)t * 16 * 768 + ko, 768, lane);
#pragma unroll
      for (int t = 0; t < 4; ++t) acc[t] = wmma_bf16(a_cur, w_cur[t], acc[t]);
      a_cur = a_nxt;
#pragma unroll
      for (int t = 0; t < 4; ++t) w_cur[t] = w_nxt[t];
    }
#pragma unroll
    for (int t = 0; t < 4; ++t) acc[t] = wmma_bf16(a_cur, w_cur[t], acc[t]);
  }

  const int col = lane & 15;
  const int rsel = (lane >> 4) * 8;
  const size_t bh = (size_t)b * 8 + h;
#pragma unroll
  for (int t = 0; t < 4; ++t) {
#pragma unroll
    for (int i = 0; i < 8; ++i) {
      int m = m0 + rsel + i;
      int dh = t * 16 + col;
      unsigned short val = (m < 77) ? f32_to_bf16_bits(acc[t][i]) : (unsigned short)0;
      if (isV) VTbuf[(bh * 64 + dh) * 96 + m] = val;
      else     Kbuf[(bh * 96 + m) * 64 + dh] = val;
    }
  }
}

// ---------------------------------------------------------------------------
// Attention: per wave one (b, h, 16-row q tile).
//   S = Q K^T (6 m-tiles x 2 k-steps), masked softmax over m (77 of 96),
//   P via LDS (layout change to A-operand), O = P V (4 dh-tiles x 3 k-steps).
// All K fragments preloaded before score WMMAs; V fragments issued before the
// softmax VALU work so global latency overlaps the reductions.
// grid = B*H*256/8 = 2048 blocks of 8 waves; 24 KB LDS per block.
// ---------------------------------------------------------------------------
__global__ __launch_bounds__(256)
void attention_kernel(const unsigned short* __restrict__ qb,     // [B*4096][512]
                      const unsigned short* __restrict__ Kbuf,   // [B*H][96][64]
                      const unsigned short* __restrict__ VTbuf,  // [B*H][64][96]
                      unsigned short* __restrict__ ob) {         // [B*4096][512]
  __shared__ __align__(16) unsigned short pshare[8][16][96];
  const int lane = threadIdx.x & 31;
  const int wave = threadIdx.x >> 5;
  const int task = blockIdx.x * 8 + wave;
  const int nt = task & 255;
  const int h = (task >> 8) & 7;
  const int b = task >> 11;
  const int n0 = nt * 16;
  const size_t bh = (size_t)b * 8 + h;
  const size_t qrow = (size_t)b * 4096 + n0;

  // Q fragments: 16 rows x 64 dh -> two K=32 steps, reused across all m-tiles.
  v16bf qf[2];
#pragma unroll
  for (int kk = 0; kk < 2; ++kk)
    qf[kk] = load_frag_bf16(qb + qrow * 512 + h * 64 + kk * 32, 512, lane);

  // Preload all K fragments, then run the 12 score WMMAs back to back.
  // B operand = K^T, i.e. K rows load exactly like an A fragment.
  v16bf kf[6][2];
#pragma unroll
  for (int t = 0; t < 6; ++t)
#pragma unroll
    for (int kk = 0; kk < 2; ++kk)
      kf[t][kk] = load_frag_bf16(Kbuf + (bh * 96 + t * 16) * 64 + kk * 32, 64, lane);

  v8f s[6];
#pragma unroll
  for (int t = 0; t < 6; ++t) {
    v8f acc = {};
    acc = wmma_bf16(qf[0], kf[t][0], acc);
    acc = wmma_bf16(qf[1], kf[t][1], acc);
    s[t] = acc;
  }

  // Issue V fragment loads now; they only depend on (b,h) and overlap softmax.
  // B operand = V (96x64); B^T = V^T stored row-major -> A-style load.
  v16bf vf[4][3];
#pragma unroll
  for (int t = 0; t < 4; ++t)
#pragma unroll
    for (int kk = 0; kk < 3; ++kk)
      vf[t][kk] = load_frag_bf16(VTbuf + (bh * 64 + t * 16) * 96 + kk * 32, 96, lane);

  // Masked softmax over m. In the C/D tile, lane = m column, VGPR slot = row,
  // so a row reduction is a 16-lane butterfly (masks 1,2,4,8 stay in-half).
  const int mcol = lane & 15;
  const float NEG = -3.0e38f;
  const float cexp = 0.125f * 1.44269504088896f;  // DH^-0.5 * log2(e)
  float p[6][8];
#pragma unroll
  for (int i = 0; i < 8; ++i) {
    float mx = NEG;
#pragma unroll
    for (int t = 0; t < 6; ++t) {
      float v = (t * 16 + mcol < 77) ? s[t][i] : NEG;
      p[t][i] = v;
      mx = fmaxf(mx, v);
    }
#pragma unroll
    for (int off = 1; off <= 8; off <<= 1) mx = fmaxf(mx, __shfl_xor(mx, off, 32));
    float sum = 0.0f;
#pragma unroll
    for (int t = 0; t < 6; ++t) {
      float e = exp2f((p[t][i] - mx) * cexp);
      p[t][i] = e;
      sum += e;
    }
#pragma unroll
    for (int off = 1; off <= 8; off <<= 1) sum += __shfl_xor(sum, off, 32);
    float r = 1.0f / sum;
#pragma unroll
    for (int t = 0; t < 6; ++t) p[t][i] *= r;
  }

  // Bounce P through per-wave LDS slab to reshape into A-operand layout.
  const int rsel = (lane >> 4) * 8;
#pragma unroll
  for (int t = 0; t < 6; ++t)
#pragma unroll
    for (int i = 0; i < 8; ++i)
      pshare[wave][rsel + i][t * 16 + mcol] = f32_to_bf16_bits(p[t][i]);

  asm volatile("s_wait_dscnt 0" ::: "memory");  // wave-local LDS RAW fence

  v16bf pf[3];
#pragma unroll
  for (int kk = 0; kk < 3; ++kk)
    pf[kk] = load_frag_bf16(&pshare[wave][0][0] + kk * 32, 96, lane);

  v8f o[4] = {};
#pragma unroll
  for (int t = 0; t < 4; ++t)
#pragma unroll
    for (int kk = 0; kk < 3; ++kk)
      o[t] = wmma_bf16(pf[kk], vf[t][kk], o[t]);

#pragma unroll
  for (int t = 0; t < 4; ++t)
#pragma unroll
    for (int i = 0; i < 8; ++i)
      ob[(qrow + rsel + i) * 512 + h * 64 + t * 16 + mcol] = f32_to_bf16_bits(o[t][i]);
}

// ---------------------------------------------------------------------------
extern "C" void kernel_launch(void* const* d_in, const int* in_sizes, int n_in,
                              void* d_out, int out_size, void* d_ws, size_t ws_size,
                              hipStream_t stream) {
  (void)in_sizes; (void)n_in; (void)out_size; (void)ws_size;
  const float* x   = (const float*)d_in[0];  // [8][4096][512]
  const float* ctx = (const float*)d_in[1];  // [8][77][768]
  const float* Wq  = (const float*)d_in[2];  // [512][512]
  const float* Wk  = (const float*)d_in[3];  // [768][512]
  const float* Wv  = (const float*)d_in[4];  // [768][512]
  const float* Wo  = (const float*)d_in[5];  // [512][512]
  const float* bo  = (const float*)d_in[6];  // [512]
  float* out = (float*)d_out;                // [8][4096][512] fp32

  char* ws = (char*)d_ws;
  size_t off = 0;
  auto take = [&](size_t bytes) {
    char* p = ws + off;
    off = (off + bytes + 255) & ~(size_t)255;
    return p;
  };
  unsigned short* WqT = (unsigned short*)take((size_t)512 * 512 * 2);
  unsigned short* WkT = (unsigned short*)take((size_t)512 * 768 * 2);
  unsigned short* WvT = (unsigned short*)take((size_t)512 * 768 * 2);
  unsigned short* WoT = (unsigned short*)take((size_t)512 * 512 * 2);
  unsigned short* qb  = (unsigned short*)take((size_t)8 * 4096 * 512 * 2);
  unsigned short* Kb  = (unsigned short*)take((size_t)64 * 96 * 64 * 2);
  unsigned short* VTb = (unsigned short*)take((size_t)64 * 64 * 96 * 2);
  // xb (bf16 x) aliases ob (attention output): xb dies after the Q GEMM,
  // ob is first written by the attention kernel afterwards.
  unsigned short* ob  = (unsigned short*)take((size_t)8 * 4096 * 512 * 2);
  unsigned short* xb  = ob;

  // 1) weights -> bf16, transposed; x -> bf16 (memory-bound pass)
  convert_transpose_kernel<<<(512 * 512 + 255) / 256, 256, 0, stream>>>(Wq, WqT, 512, 512);
  convert_transpose_kernel<<<(768 * 512 + 255) / 256, 256, 0, stream>>>(Wk, WkT, 768, 512);
  convert_transpose_kernel<<<(768 * 512 + 255) / 256, 256, 0, stream>>>(Wv, WvT, 768, 512);
  convert_transpose_kernel<<<(512 * 512 + 255) / 256, 256, 0, stream>>>(Wo, WoT, 512, 512);
  f32_to_bf16_bulk_kernel<<<8192, 256, 0, stream>>>(x, (unsigned int*)xb,
                                                    8 * 4096 * 512 / 8);

  // 2) Q = x @ Wq  (bf16 in, bf16 out)
  gemm_bf16_wt<false, true><<<dim3(128, 8), 256, 0, stream>>>(xb, WqT, nullptr, qb);

  // 3) K,V projections into padded per-head layouts
  kv_proj_kernel<<<dim3(48, 2), 256, 0, stream>>>(ctx, WkT, WvT, Kb, VTb);

  // 4) attention core
  attention_kernel<<<2048, 256, 0, stream>>>(qb, Kb, VTb, ob);

  // 5) out = O @ Wo + bo  (fp32 output)
  gemm_bf16_wt<true, false><<<dim3(128, 8), 256, 0, stream>>>(ob, WoT, bo, out);
}
